// SingleHeadGATLayer_45595372815169
// MI455X (gfx1250) — matrix operand order repaired and checked by reading, not compiled, not run on hardware
//
#include <hip/hip_runtime.h>

// Problem constants (match reference)
#define NN 65536        // N_NODES
#define NE 1048576      // N_EDGES
#define D  64           // DIM
#define KH 3            // K_HOPS

typedef __attribute__((ext_vector_type(16))) _Float16 v16h;
typedef __attribute__((ext_vector_type(8)))  float    v8f;

// ---- monotonic float <-> uint key (for atomic max on floats) ----
__device__ __forceinline__ unsigned f2key(float f) {
  unsigned u = __float_as_uint(f);
  return (u & 0x80000000u) ? ~u : (u | 0x80000000u);
}
__device__ __forceinline__ float key2f(unsigned k) {
  unsigned u = (k & 0x80000000u) ? (k & 0x7FFFFFFFu) : ~k;
  return __uint_as_float(u);
}

// out = temp[0] * features
__global__ void gat_init_out(const float* __restrict__ feats,
                             const float* __restrict__ temp,
                             float* __restrict__ out) {
  int i = blockIdx.x * blockDim.x + threadIdx.x;
  if (i < NN * D) out[i] = temp[0] * feats[i];
}

// per-hop node-state init: m-key = 0 (< key(-inf)), denom = 0
__global__ void gat_init_node(unsigned* __restrict__ mkey,
                              float* __restrict__ denom) {
  int i = blockIdx.x * blockDim.x + threadIdx.x;
  if (i < NN) { mkey[i] = 0u; denom[i] = 0.0f; }
}

__global__ void gat_zero_vec(float* __restrict__ p, int n) {
  int i = blockIdx.x * blockDim.x + threadIdx.x;
  if (i < n) p[i] = 0.0f;
}

// -------- node scores via WMMA: S(16x2) = F_tile(16x64) @ W(64x2) --------
// One wave32 per 16-node tile. K=64 done as two v_wmma_f32_16x16x32_f16.
// B columns: N=0 -> a_src (attn_w[0:64]), N=1 -> a_dst (attn_w[64:128]); rest 0.
__global__ void gat_scores_wmma(const float* __restrict__ feats,
                                const float* __restrict__ attn_w,
                                float* __restrict__ s_src,
                                float* __restrict__ s_dst) {
  const int lane = threadIdx.x & 31;
  const int wave = (blockIdx.x * blockDim.x + threadIdx.x) >> 5;
  const int n    = lane & 15;   // N index (B/D); row-in-tile (A)
  const int hi   = lane >> 4;   // half-wave selector
  const int rowBase = wave * 16;

  v8f acc = {};
  #pragma unroll
  for (int c = 0; c < 2; ++c) {           // K chunks of 32
    // A layout (16-bit 16x32): lane<16 holds M=lane, K = {0..7, 16..23};
    // lane>=16 holds M=lane-16, K = {8..15, 24..31}.
    v16h a;
    const float* frow = feats + (rowBase + n) * D + c * 32;
    const int kb = hi * 8;
    #pragma unroll
    for (int j = 0; j < 8; ++j) {
      a[j]     = (_Float16)frow[kb + j];
      a[8 + j] = (_Float16)frow[16 + kb + j];
    }
    // B layout (16-bit 32x16): lanes 0-15 -> N=lane, K={0..7,16..23}; lanes 16-31 -> K={8..15,24..31}
    v16h b = {};
    if (n < 2) {
      const float* wcol = attn_w + n * D + c * 32;
      #pragma unroll
      for (int j = 0; j < 8; ++j) {
        b[j]     = (_Float16)wcol[kb + j];
        b[8 + j] = (_Float16)wcol[16 + kb + j];
      }
    }
    acc = __builtin_amdgcn_wmma_f32_16x16x32_f16(false, a, false, b,
                                                 (short)0, acc, false, false);
  }
  // D layout: acc[r]: lanes 0-15 -> (M=r, N=lane); lanes 16-31 -> (M=8+r, N=lane-16)
  if (n < 2) {
    float* outp = (n == 0) ? s_src : s_dst;
    const int m0 = rowBase + hi * 8;
    #pragma unroll
    for (int r = 0; r < 8; ++r) outp[m0 + r] = acc[r];
  }
}

// e = leaky_relu(s_src[src] + s_dst[dst] + b); store; atomic segment max on dst
__global__ void gat_edge_score(const int* __restrict__ src,
                               const int* __restrict__ dst,
                               const float* __restrict__ s_src,
                               const float* __restrict__ s_dst,
                               const float* __restrict__ attn_b,
                               float* __restrict__ ebuf,
                               unsigned* __restrict__ mkey) {
  int i = blockIdx.x * blockDim.x + threadIdx.x;
  if (i >= NE) return;
  int s = src[i], t = dst[i];
  float e = s_src[s] + s_dst[t] + attn_b[0];
  e = (e > 0.0f) ? e : 0.01f * e;          // jax.nn.leaky_relu slope 0.01
  ebuf[i] = e;
  atomicMax(&mkey[t], f2key(e));
}

// ex = exp(e - m[dst]); store in place; atomic segment sum into denom
__global__ void gat_edge_exp(const int* __restrict__ dst,
                             const unsigned* __restrict__ mkey,
                             float* __restrict__ ebuf,
                             float* __restrict__ denom) {
  int i = blockIdx.x * blockDim.x + threadIdx.x;
  if (i >= NE) return;
  int t = dst[i];
  float ex = expf(ebuf[i] - key2f(mkey[t]));
  ebuf[i] = ex;
  atomicAdd(&denom[t], ex);
}

// next[dst] += (ex/denom[dst]) * cur[src] ; 16 lanes per edge, float4 each
__global__ void gat_aggregate(const int* __restrict__ src,
                              const int* __restrict__ dst,
                              const float* __restrict__ ebuf,
                              const float* __restrict__ denom,
                              const float* __restrict__ cur,
                              float* __restrict__ nxt) {
  int tid  = blockIdx.x * blockDim.x + threadIdx.x;
  int edge = tid >> 4;
  int q    = tid & 15;
  if (edge >= NE) return;
  int s = src[edge], t = dst[edge];
  float w = ebuf[edge] / denom[t];
  const float4 v = *(const float4*)(cur + s * D + q * 4);
  float* o = nxt + t * D + q * 4;
  atomicAdd(o + 0, w * v.x);
  atomicAdd(o + 1, w * v.y);
  atomicAdd(o + 2, w * v.z);
  atomicAdd(o + 3, w * v.w);
}

// out += temp[kidx] * buf
__global__ void gat_accum_out(const float* __restrict__ buf,
                              const float* __restrict__ temp, int kidx,
                              float* __restrict__ out) {
  int i = blockIdx.x * blockDim.x + threadIdx.x;
  if (i < NN * D) out[i] += temp[kidx] * buf[i];
}

extern "C" void kernel_launch(void* const* d_in, const int* in_sizes, int n_in,
                              void* d_out, int out_size, void* d_ws, size_t ws_size,
                              hipStream_t stream) {
  const float* features = (const float*)d_in[0];
  const int*   src      = (const int*)d_in[1];
  const int*   dst      = (const int*)d_in[2];
  const float* attn_w   = (const float*)d_in[3];
  const float* attn_b   = (const float*)d_in[4];
  const float* temp     = (const float*)d_in[5];
  float*       out      = (float*)d_out;

  // workspace layout (floats)
  float*    ws    = (float*)d_ws;
  float*    s_src = ws;                       // NN
  float*    s_dst = s_src + NN;               // NN
  unsigned* mkey  = (unsigned*)(s_dst + NN);  // NN
  float*    denom = (float*)(mkey + NN);      // NN
  float*    ebuf  = denom + NN;               // NE
  float*    buf0  = ebuf + NE;                // NN*D
  float*    buf1  = buf0 + NN * D;            // NN*D
  size_t needed = (size_t)(4 * NN + NE + 2 * NN * D) * sizeof(float);
  if (ws_size < needed) return;               // refuse to scribble OOB

  const int TB = 256;
  const int gNodeVec = (NN * D) / TB;         // 16384
  const int gNode    = NN / TB;               // 256
  const int gEdge    = NE / TB;               // 4096
  const int gAgg     = (NE * 16) / TB;        // 65536
  const int gScore   = (NN / 16) * 32 / TB;   // 512 (one wave32 per 16 nodes)

  gat_init_out<<<gNodeVec, TB, 0, stream>>>(features, temp, out);

  const float* cur = features;
  float* nxt = buf0;
  for (int k = 0; k < KH; ++k) {
    gat_init_node<<<gNode, TB, 0, stream>>>(mkey, denom);
    gat_zero_vec<<<gNodeVec, TB, 0, stream>>>(nxt, NN * D);
    gat_scores_wmma<<<gScore, TB, 0, stream>>>(cur, attn_w, s_src, s_dst);
    gat_edge_score<<<gEdge, TB, 0, stream>>>(src, dst, s_src, s_dst, attn_b, ebuf, mkey);
    gat_edge_exp<<<gEdge, TB, 0, stream>>>(dst, mkey, ebuf, denom);
    gat_aggregate<<<gAgg, TB, 0, stream>>>(src, dst, ebuf, denom, cur, nxt);
    gat_accum_out<<<gNodeVec, TB, 0, stream>>>(nxt, temp, k + 1, out);
    cur = nxt;
    nxt = (nxt == buf0) ? buf1 : buf0;
  }
}
